// LSTMModel_86612310491805
// MI455X (gfx1250) — matrix operand order repaired and checked by reading, not compiled, not run on hardware
//
#include <hip/hip_runtime.h>
#include <hip/hip_bf16.h>
#include <math.h>

// ---------------------------------------------------------------------------
// 3-layer LSTM (B=256, T=1024, H=150) + FC, CDNA5 WMMA implementation.
//   H padded 150->160 (HP), 4H padded 600->640 (NG). f16 WMMA, f32 accum.
// Strategy:
//  * input projections = parallel WMMA GEMMs over B*T rows (L2-resident wts)
//  * recurrence = 16 persistent blocks (M=16 batch rows each); w_hh (200KB
//    f16), h, c, gate preacts all LDS-resident (CDNA5: 320KB LDS per WG).
//    200 v_wmma_f32_16x16x32_f16 per block per timestep, zero global sync.
//  * B-fragments preloaded per tile into distinct registers so loads clause
//    and pipeline against the WMMA accumulator chain (no loadcnt-0 stalls).
// ---------------------------------------------------------------------------

#define BATCH 256
#define TLEN  1024
#define HREAL 150
#define HP    160          // padded hidden (5 * K32)
#define NG    640          // padded gate dim = 4*HP (40 N-tiles of 16)
#define WPER  (NG * HP)    // padded weight elements per matrix = 102400

typedef _Float16 v16h __attribute__((ext_vector_type(16)));
typedef _Float16 v8h  __attribute__((ext_vector_type(8)));
typedef float    v8f  __attribute__((ext_vector_type(8)));

__device__ __forceinline__ v16h cat8(v8h a, v8h b) {
    return __builtin_shufflevector(a, b, 0,1,2,3,4,5,6,7,8,9,10,11,12,13,14,15);
}

__device__ __forceinline__ float sigm(float x) {
    return 1.0f / (1.0f + __expf(-x));
}

// --- pad/convert weights to f16 [NG][HP]; fuse biases to f32 [NG] per layer
__global__ __launch_bounds__(256)
void prep_w(const float* __restrict__ whh1, const float* __restrict__ whh2,
            const float* __restrict__ whh3, const float* __restrict__ wih2,
            const float* __restrict__ wih3,
            const float* __restrict__ bih1, const float* __restrict__ bhh1,
            const float* __restrict__ bih2, const float* __restrict__ bhh2,
            const float* __restrict__ bih3, const float* __restrict__ bhh3,
            _Float16* __restrict__ wph,   // 3 * WPER  (w_hh layers 1..3)
            _Float16* __restrict__ wpi,   // 2 * WPER  (w_ih layers 2..3)
            float* __restrict__ bias)     // 3 * NG
{
    int idx = blockIdx.x * 256 + threadIdx.x;
    if (idx < 5 * WPER) {
        int l = idx / WPER, r = idx - l * WPER;
        int n = r / HP, k = r - n * HP;
        int g = n / HP, j = n - g * HP;
        const float* src = (l == 0) ? whh1 : (l == 1) ? whh2 : (l == 2) ? whh3
                                                   : (l == 3) ? wih2 : wih3;
        float v = 0.0f;
        if (j < HREAL && k < HREAL) v = src[(g * HREAL + j) * HREAL + k];
        _Float16* dst = (l < 3) ? (wph + l * WPER) : (wpi + (l - 3) * WPER);
        dst[r] = (_Float16)v;
    } else if (idx < 5 * WPER + 3 * NG) {
        int r = idx - 5 * WPER;
        int l = r / NG, n = r - l * NG;
        int g = n / HP, j = n - g * HP;
        float v = 0.0f;
        if (j < HREAL) {
            int o = g * HREAL + j;
            v = (l == 0) ? (bih1[o] + bhh1[o])
              : (l == 1) ? (bih2[o] + bhh2[o])
                         : (bih3[o] + bhh3[o]);
        }
        bias[l * NG + n] = v;
    }
}

// --- layer-1 input projection (IN=1): xg[bt][n] = x[bt]*w_ih1[n'] + bias
__global__ __launch_bounds__(256)
void xg_layer1(const float* __restrict__ x, const float* __restrict__ wih1,
               const float* __restrict__ bih1, const float* __restrict__ bhh1,
               float* __restrict__ xg)
{
    size_t idx = (size_t)blockIdx.x * 256 + threadIdx.x;
    if (idx >= (size_t)BATCH * TLEN * NG) return;
    int n = (int)(idx % NG);
    size_t bt = idx / NG;
    int g = n / HP, j = n - g * HP;
    float v = 0.0f;
    if (j < HREAL) {
        int o = g * HREAL + j;
        v = x[bt] * wih1[o] + bih1[o] + bhh1[o];   // w_ih1 is [600][1]
    }
    xg[idx] = v;
}

// --- input projection GEMM (layers 2,3): M=B*T (16/block), N=640, K=160
__global__ __launch_bounds__(256)
void gemm_xg(const _Float16* __restrict__ hin,  // [B*T][HP]
             const _Float16* __restrict__ wip,  // [NG][HP] padded f16
             const float* __restrict__ bias,    // [NG]
             float* __restrict__ xg)            // [B*T][NG]
{
    const int bt0  = blockIdx.x * 16;
    const int lane = threadIdx.x & 31;
    const int wv   = threadIdx.x >> 5;
    const int kh   = lane >> 4;     // K-half select
    const int nl   = lane & 15;     // M (A) / N (B,D) lane index

    // A fragments: row m = bt0+nl, K-half kh, chunks at K+0 and K+16
    v16h afrag[5];
#pragma unroll
    for (int kt = 0; kt < 5; ++kt) {
        const _Float16* pa = &hin[(size_t)(bt0 + nl) * HP + kt * 32 + kh * 8];
        afrag[kt] = cat8(*(const v8h*)pa, *(const v8h*)(pa + 16));
    }
#pragma unroll
    for (int i = 0; i < 5; ++i) {
        const int n0 = (wv * 5 + i) * 16;
        const _Float16* pbrow = &wip[(size_t)(n0 + nl) * HP + kh * 8];
        // Preload all 5 B fragments into distinct registers: the 10 b128
        // loads clause together and pipeline against the WMMA chain.
        v16h bfr[5];
#pragma unroll
        for (int kt = 0; kt < 5; ++kt)
            bfr[kt] = cat8(*(const v8h*)(pbrow + kt * 32), *(const v8h*)(pbrow + kt * 32 + 16));
        const float bv = bias[n0 + nl];
        v8f acc;
#pragma unroll
        for (int r = 0; r < 8; ++r) acc[r] = bv;
#pragma unroll
        for (int kt = 0; kt < 5; ++kt)
            acc = __builtin_amdgcn_wmma_f32_16x16x32_f16(
                false, afrag[kt], false, bfr[kt], (short)0, acc, false, false);
#pragma unroll
        for (int r = 0; r < 8; ++r)
            xg[(size_t)(bt0 + r + kh * 8) * NG + n0 + nl] = acc[r];
    }
}

// --- persistent recurrence: one block per 16 batch rows, full time loop
__global__ __launch_bounds__(256)
void lstm_rec(const float* __restrict__ xg,     // [B][T][NG] f32 (bias fused)
              const _Float16* __restrict__ wp,  // [NG][HP] padded w_hh f16
              _Float16* __restrict__ hout)      // [B*T][HP] f16
{
    __shared__ _Float16 wsh[NG * HP];    // 200 KB: recurrent weights
    __shared__ _Float16 hsh[16 * HP];    //   5 KB: h state (WMMA A source)
    __shared__ float    csh[16 * HP];    //  10 KB: c state
    __shared__ float    gsh[16 * NG];    //  40 KB: gate preactivations

    const int m0  = blockIdx.x * 16;
    const int tid = threadIdx.x;

    for (int i = tid * 8; i < NG * HP; i += 256 * 8)
        *(v8h*)&wsh[i] = *(const v8h*)&wp[i];
    for (int i = tid; i < 16 * HP; i += 256) {
        hsh[i] = (_Float16)0.0f;
        csh[i] = 0.0f;
    }
    __syncthreads();

    const int lane = tid & 31;
    const int wv   = tid >> 5;
    const int kh   = lane >> 4;
    const int nl   = lane & 15;

    for (int t = 0; t < TLEN; ++t) {
        // A fragments from LDS-resident h (row m = nl).
        v16h afrag[5];
#pragma unroll
        for (int kt = 0; kt < 5; ++kt) {
            const _Float16* pa = &hsh[nl * HP + kt * 32 + kh * 8];
            afrag[kt] = cat8(*(const v8h*)pa, *(const v8h*)(pa + 16));
        }
        // 5 N-tiles per wave: preact = xg (C init) + h @ w_hh^T (WMMA).
#pragma unroll
        for (int i = 0; i < 5; ++i) {
            const int n0   = (wv * 5 + i) * 16;
            const int gcol = n0 + nl;
            // Issue the scattered xg global loads first so their latency
            // hides under the LDS B-fragment reads and the WMMA chain.
            const float* px =
                &xg[((size_t)(m0 + kh * 8) * TLEN + t) * NG + gcol];
            v8f acc;
#pragma unroll
            for (int r = 0; r < 8; ++r)
                acc[r] = px[(size_t)r * TLEN * NG];
            if (i == 0 && t + 1 < TLEN)
                __builtin_prefetch(px + NG, 0, 0);   // next timestep row
            // Preload all 5 LDS B fragments (10 ds_load_b128) before WMMAs.
            const _Float16* pbrow = &wsh[(n0 + nl) * HP + kh * 8];
            v16h bfr[5];
#pragma unroll
            for (int kt = 0; kt < 5; ++kt)
                bfr[kt] = cat8(*(const v8h*)(pbrow + kt * 32), *(const v8h*)(pbrow + kt * 32 + 16));
#pragma unroll
            for (int kt = 0; kt < 5; ++kt)
                acc = __builtin_amdgcn_wmma_f32_16x16x32_f16(
                    false, afrag[kt], false, bfr[kt], (short)0, acc, false, false);
#pragma unroll
            for (int r = 0; r < 8; ++r)
                gsh[(r + kh * 8) * NG + gcol] = acc[r];
        }
        __syncthreads();

        // Gate nonlinearities + state update (pytorch order i,f,g,o).
        for (int e = tid; e < 16 * HP; e += 256) {
            const int m = e / HP;
            const int j = e - m * HP;
            const float* gr = &gsh[m * NG];
            float gi = gr[j];
            float gf = gr[HP + j];
            float gg = gr[2 * HP + j];
            float go = gr[3 * HP + j];
            float c  = csh[e];
            c = sigm(gf) * c + sigm(gi) * tanhf(gg);
            float h = sigm(go) * tanhf(c);
            csh[e] = c;
            hsh[e] = (_Float16)h;
            hout[((size_t)(m0 + m) * TLEN + t) * HP + j] = (_Float16)h;
        }
        __syncthreads();
    }
}

// --- final FC: out[bt] = fc_b + sum_{j<150} h3[bt][j] * fc_w[j]
__global__ __launch_bounds__(256)
void fc_k(const _Float16* __restrict__ h, const float* __restrict__ fw,
          const float* __restrict__ fb, float* __restrict__ out)
{
    int bt = blockIdx.x * 256 + threadIdx.x;
    if (bt >= BATCH * TLEN) return;
    const _Float16* hp = &h[(size_t)bt * HP];
    float s = fb[0];
#pragma unroll 5
    for (int j = 0; j < HREAL; ++j) s += (float)hp[j] * fw[j];
    out[bt] = s;
}

// ---------------------------------------------------------------------------
extern "C" void kernel_launch(void* const* d_in, const int* in_sizes, int n_in,
                              void* d_out, int out_size, void* d_ws, size_t ws_size,
                              hipStream_t stream) {
    (void)in_sizes; (void)n_in; (void)out_size; (void)ws_size;
    const float* x     = (const float*)d_in[0];
    const float* wih1  = (const float*)d_in[1];
    const float* whh1  = (const float*)d_in[2];
    const float* bih1  = (const float*)d_in[3];
    const float* bhh1  = (const float*)d_in[4];
    const float* wih2  = (const float*)d_in[5];
    const float* whh2  = (const float*)d_in[6];
    const float* bih2  = (const float*)d_in[7];
    const float* bhh2  = (const float*)d_in[8];
    const float* wih3  = (const float*)d_in[9];
    const float* whh3  = (const float*)d_in[10];
    const float* bih3  = (const float*)d_in[11];
    const float* bhh3  = (const float*)d_in[12];
    const float* fcw   = (const float*)d_in[13];
    const float* fcb   = (const float*)d_in[14];
    float* out = (float*)d_out;

    // Workspace carve-up (offsets all 256B-aligned; total ~801 MB).
    char* p = (char*)d_ws;
    _Float16* wph  = (_Float16*)p; p += (size_t)3 * WPER * sizeof(_Float16);
    _Float16* wpi  = (_Float16*)p; p += (size_t)2 * WPER * sizeof(_Float16);
    float*    bias = (float*)p;    p += (size_t)3 * NG * sizeof(float);
    float*    xg   = (float*)p;    p += (size_t)BATCH * TLEN * NG * sizeof(float);
    _Float16* h1   = (_Float16*)p; p += (size_t)BATCH * TLEN * HP * sizeof(_Float16);
    _Float16* h2   = (_Float16*)p;

    const size_t BT = (size_t)BATCH * TLEN;

    {
        int total = 5 * WPER + 3 * NG;
        prep_w<<<(total + 255) / 256, 256, 0, stream>>>(
            whh1, whh2, whh3, wih2, wih3,
            bih1, bhh1, bih2, bhh2, bih3, bhh3, wph, wpi, bias);
    }
    // Layer 1
    {
        size_t total = BT * NG;
        xg_layer1<<<(unsigned)((total + 255) / 256), 256, 0, stream>>>(
            x, wih1, bih1, bhh1, xg);
        lstm_rec<<<BATCH / 16, 256, 0, stream>>>(xg, wph, h1);
    }
    // Layer 2
    gemm_xg<<<(unsigned)(BT / 16), 256, 0, stream>>>(h1, wpi, bias + NG, xg);
    lstm_rec<<<BATCH / 16, 256, 0, stream>>>(xg, wph + WPER, h2);
    // Layer 3
    gemm_xg<<<(unsigned)(BT / 16), 256, 0, stream>>>(h2, wpi + WPER, bias + 2 * NG, xg);
    lstm_rec<<<BATCH / 16, 256, 0, stream>>>(xg, wph + 2 * WPER, h1);
    // FC
    fc_k<<<(unsigned)((BT + 255) / 256), 256, 0, stream>>>(h1, fcw, fcb, out);
}
